// GeoUmpForPretrain_32693291057152
// MI455X (gfx1250) — matrix operand order repaired and checked by reading, not compiled, not run on hardware
//
#include <hip/hip_runtime.h>
#include <hip/hip_bf16.h>

// ---------------------------------------------------------------------------
// MI455X / gfx1250 implementation.
// All GEMMs: v_wmma_f32_16x16x32_bf16 (bf16 in, f32 accum), wave32,
// 8 waves / block, 64x64 tiles, double-buffered LDS staging (one barrier per
// k-step). Pure-copy staging (NT GEMM, NN GEMM A-tile) uses the CDNA5 async
// global->LDS DMA path (GLOBAL_LOAD_ASYNC_TO_LDS_B128 + s_wait_asynccnt).
// ---------------------------------------------------------------------------

typedef __bf16 bf16;
typedef __attribute__((ext_vector_type(16))) __bf16 v16bf;
typedef __attribute__((ext_vector_type(8)))  __bf16 v8bf;
typedef __attribute__((ext_vector_type(8)))  float  v8f;
typedef __attribute__((address_space(3)))    bf16   lds_bf16;

#define B_  16
#define S_  512
#define H_  768
#define T_  2
#define N_  32
#define L_  64
#define E_  8
#define D_  128
#define A_  128

__device__ __forceinline__ v8f wmma_bf16(v16bf a, v16bf b, v8f c) {
  return __builtin_amdgcn_wmma_f32_16x16x32_bf16(
      false, a, false, b, (short)0, c, false, false);
}

__device__ __forceinline__ v16bf join16(v8bf lo, v8bf hi) {
  return __builtin_shufflevector(lo, hi, 0, 1, 2, 3, 4, 5, 6, 7,
                                 8, 9, 10, 11, 12, 13, 14, 15);
}

// one v_cvt_pk_bf16_f32: two fp32 -> packed 2xbf16 dword
__device__ __forceinline__ unsigned pk2(float lo, float hi) {
  union { bf16 h[2]; unsigned u; } t;
  t.h[0] = (bf16)lo;
  t.h[1] = (bf16)hi;
  return t.u;
}

// CDNA5 async DMA: copy 16B/lane global -> LDS, tracked by ASYNCcnt.
__device__ __forceinline__ void async_g2l_b128(bf16* ldst, const bf16* gsrc) {
  unsigned lds = (unsigned)(size_t)(lds_bf16*)ldst;  // 32-bit LDS address
  asm volatile("global_load_async_to_lds_b128 %0, %1, off"
               :: "v"(lds), "v"(gsrc) : "memory");
}
__device__ __forceinline__ void wait_async0() {
  asm volatile("s_wait_asynccnt 0x0" ::: "memory");
}

// A fragment (16x32) from LDS tile laid [row][k], stride LD elements.
// lanes 0-15 -> K {0..15}; lanes 16-31 -> K {16..31} (matches B fragments).
template <int LD>
__device__ __forceinline__ v16bf load_a_frag(const bf16* s, int m0, int k0, int lane) {
  const bf16* p = s + (m0 + (lane & 15)) * LD + k0 + ((lane >> 4) << 4);
  return join16(*(const v8bf*)p, *(const v8bf*)(p + 8));
}

// B fragment (32x16) from a TRANSPOSED tile laid [n][k], stride LD elements.
template <int LD>
__device__ __forceinline__ v16bf load_bt_frag_lds(const bf16* s, int n0, int k0, int lane) {
  const bf16* p = s + (n0 + (lane & 15)) * LD + k0 + ((lane >> 4) << 4);
  return join16(*(const v8bf*)p, *(const v8bf*)(p + 8));
}

// Same but from global memory (row-major [n][k], k contiguous).
__device__ __forceinline__ v16bf load_bt_frag_g(const bf16* g, int ld, int n0, int k0, int lane) {
  const bf16* p = g + (long)(n0 + (lane & 15)) * ld + k0 + ((lane >> 4) << 4);
  return join16(*(const v8bf*)p, *(const v8bf*)(p + 8));
}

// ---------------------------------------------------------------------------
// Elementwise / gather helpers
// ---------------------------------------------------------------------------

__global__ void cvt_f32_bf16(const float* __restrict__ src, bf16* __restrict__ dst, int n) {
  int i = blockIdx.x * 256 + threadIdx.x;
  if (i < n) dst[i] = (bf16)src[i];
}

__global__ void copy_f32(const float* __restrict__ src, float* __restrict__ dst, long n) {
  long i = (long)blockIdx.x * 256 + threadIdx.x;
  if (i < n) dst[i] = src[i];
}

__global__ void rowidx_kernel(const int* __restrict__ spans, int* __restrict__ rowidx) {
  int i = threadIdx.x;  // 256 = B*T*E
  int b = i >> 4, t = (i >> 3) & 1, e = i & 7;
  rowidx[i] = b * S_ + spans[b * T_ + t] + e;
}

__global__ void feas_kernel(const float* __restrict__ xs, const int* __restrict__ spans,
                            float* __restrict__ feas) {
  int bt = blockIdx.x;  // 32
  int b  = bt >> 1;
  int sp = spans[bt];
  for (int h = threadIdx.x; h < H_; h += 256) {
    float s = 0.f;
#pragma unroll
    for (int e = 0; e < E_; ++e) s += xs[((long)b * S_ + sp + e) * H_ + h];
    feas[bt * H_ + h] = s * (1.f / E_);
  }
}

// ---------------------------------------------------------------------------
// Generic GEMM:  C[M,N] = gather(A_f32)[M,K] * Bw_bf16[K,N] + bias
// K % 32 == 0, N % 64 == 0 at all call sites. Double-buffered pipeline.
// Optional At: write the staged (converted) A tiles transposed to global as
// At[row/64][k][row%64] bf16 -- used to produce neighT during k-projection.
// ---------------------------------------------------------------------------

__global__ __launch_bounds__(256) void gemm_a32_wb(
    const float* __restrict__ A, const int* __restrict__ rowidx,
    int M, int K, int N,
    const bf16* __restrict__ Bw, const float* __restrict__ bias,
    float* __restrict__ Cf, bf16* __restrict__ Cb, bf16* __restrict__ At) {
  __shared__ __align__(16) bf16 sA[2][64 * 40];   // [row][k]
  __shared__ __align__(16) bf16 sBt[2][64 * 40];  // [n][k]
  const int tid  = threadIdx.x;
  const int lane = tid & 31;
  const int wave = tid >> 5;
  const int gm0  = blockIdx.x * 64;
  const int gn0  = blockIdx.y * 64;
  const int mt   = (wave & 3) * 16;
  const int nt   = (wave >> 2) * 32;

  const int ar = tid >> 2, ac = (tid & 3) * 8;  // A: 4 threads/row
  const int bk = tid >> 3, bn = (tid & 7) * 8;  // B: 8 threads/k-row

  // loop-invariant per-thread source pointers
  const int  gr     = gm0 + ar;
  const bool avalid = (gr < M);
  const int  arow   = avalid ? (rowidx ? rowidx[gr] : gr) : 0;
  const float* aptr = A + (long)arow * K + ac;
  const bf16*  bptr = Bw + (long)bk * N + gn0 + bn;
  const bool  doAt  = (At != nullptr) && (blockIdx.y == 0);
  const long  atBase = (long)gm0 * K;

  auto stage = [&](int buf, int k0) {
    // A tile: two b128 global loads, 4x v_cvt_pk_bf16_f32, one b128 LDS store
    float4 u0 = *(const float4*)(aptr + k0);
    float4 u1 = *(const float4*)(aptr + k0 + 4);
    uint4 pk = {pk2(u0.x, u0.y), pk2(u0.z, u0.w), pk2(u1.x, u1.y), pk2(u1.z, u1.w)};
    if (!avalid) { pk.x = 0u; pk.y = 0u; pk.z = 0u; pk.w = 0u; }
    *(uint4*)&sA[buf][ar * 40 + ac] = pk;
    if (doAt) {  // transposed bf16 side-product (neighT); avalid always true here
      union { uint4 u; bf16 h[8]; } t;
      t.u = pk;
#pragma unroll
      for (int j = 0; j < 8; ++j) At[atBase + (long)(k0 + ac + j) * 64 + ar] = t.h[j];
    }
    // B tile transposed: one b128 read, 8 strided b16 LDS writes
    v8bf v = *(const v8bf*)(bptr + (long)k0 * N);
#pragma unroll
    for (int j = 0; j < 8; ++j) sBt[buf][(bn + j) * 40 + bk] = v[j];
    __builtin_prefetch(bptr + (long)(k0 + 64) * N, 0, 1);  // speculative
  };

  v8f acc0 = {0, 0, 0, 0, 0, 0, 0, 0};
  v8f acc1 = {0, 0, 0, 0, 0, 0, 0, 0};

  stage(0, 0);
  __syncthreads();
  const int nk = K >> 5;
  for (int s = 0; s < nk; ++s) {
    const int cur = s & 1;
    if (s + 1 < nk) stage(cur ^ 1, (s + 1) << 5);
    v16bf a  = load_a_frag<40>(sA[cur], mt, 0, lane);
    v16bf b0 = load_bt_frag_lds<40>(sBt[cur], nt, 0, lane);
    v16bf b1 = load_bt_frag_lds<40>(sBt[cur], nt + 16, 0, lane);
    acc0 = wmma_bf16(a, b0, acc0);
    acc1 = wmma_bf16(a, b1, acc1);
    __syncthreads();
  }

  const int col   = lane & 15;
  const int rbase = (lane >> 4) * 8;
#pragma unroll
  for (int i = 0; i < 8; ++i) {
    int gro = gm0 + mt + rbase + i;
    if (gro >= M) continue;
    int gc0 = gn0 + nt + col;
    int gc1 = gn0 + nt + 16 + col;
    float v0 = acc0[i] + bias[gc0];
    float v1 = acc1[i] + bias[gc1];
    if (Cf) { Cf[(long)gro * N + gc0] = v0; Cf[(long)gro * N + gc1] = v1; }
    if (Cb) { Cb[(long)gro * N + gc0] = (bf16)v0; Cb[(long)gro * N + gc1] = (bf16)v1; }
  }
}

// ---------------------------------------------------------------------------
// Fused per-neighbor SDPA: logits = q k^T / sqrt(D); softmax over L;
// msgs = attn * V; pooled = mean_E(msgs). One block per (b,t,n).
// ---------------------------------------------------------------------------

__global__ __launch_bounds__(256) void neighbor_sdpa(
    const float* __restrict__ qf,      // [B*T*E, D] fp32
    const bf16*  __restrict__ k_bf,    // [B*T*N, L, D] bf16
    const bf16*  __restrict__ neighT,  // [B*T*N, H, L] bf16
    float* __restrict__ msgs,          // [B*T*N, E, H] fp32
    float* __restrict__ pooled) {      // [B*T*N, H] fp32
  const int btn = blockIdx.x;  // 0..1023
  const int bt  = btn >> 5;
  __shared__ __align__(16) bf16  sQ[16 * 136];
  __shared__ __align__(16) float sLog[8 * 64];
  __shared__ __align__(16) bf16  sAttn[16 * 72];
  const int tid = threadIdx.x, lane = tid & 31, wave = tid >> 5;

  // stage q: 16 rows x 128 cols; rows 8..15 zero
  {
    const int r = tid >> 4;        // 0..15
    const int c = (tid & 15) * 8;  // 0..120
    uint4 pk = {0u, 0u, 0u, 0u};
    if (r < E_) {
      const float* src = qf + (long)(bt * E_ + r) * D_ + c;
      float4 u0 = *(const float4*)src;
      float4 u1 = *(const float4*)(src + 4);
      pk.x = pk2(u0.x, u0.y); pk.y = pk2(u0.z, u0.w);
      pk.z = pk2(u1.x, u1.y); pk.w = pk2(u1.z, u1.w);
    }
    *(uint4*)&sQ[r * 136 + c] = pk;
  }
  for (int idx = tid; idx < (16 * 72) / 2; idx += 256) ((uint*)sAttn)[idx] = 0u;
  __syncthreads();

  // logits: waves 0..3 each one 16-col tile of [16 x 64]
  const bf16* kb = k_bf + (long)btn * L_ * D_;
  if (wave < 4) {
    v8f acc = {0, 0, 0, 0, 0, 0, 0, 0};
    const int ntile = wave * 16;
#pragma unroll
    for (int k0 = 0; k0 < D_; k0 += 32) {
      v16bf a = load_a_frag<136>(sQ, 0, k0, lane);
      v16bf b = load_bt_frag_g(kb, D_, ntile, k0, lane);
      acc = wmma_bf16(a, b, acc);
    }
    const float scale = 0.08838834764831845f;  // 1/sqrt(128)
    if (lane < 16) {
#pragma unroll
      for (int i = 0; i < 8; ++i) sLog[i * 64 + ntile + lane] = acc[i] * scale;
    }
  }
  __syncthreads();

  // softmax over L=64 for rows 0..7
  if (tid < E_) {
    float m = -1e30f;
    for (int j = 0; j < L_; ++j) m = fmaxf(m, sLog[tid * 64 + j]);
    float s = 0.f;
    for (int j = 0; j < L_; ++j) {
      float e = __expf(sLog[tid * 64 + j] - m);
      sLog[tid * 64 + j] = e;
      s += e;
    }
    float inv = 1.f / s;
    for (int j = 0; j < L_; ++j) sAttn[tid * 72 + j] = (bf16)(sLog[tid * 64 + j] * inv);
  }
  __syncthreads();

  // msgs = attn(16x64) * V(64x768); 8 waves x 6 col-tiles
  const bf16* Vt = neighT + (long)btn * H_ * L_;
  float* mo = msgs + (long)btn * E_ * H_;
  float* po = pooled + (long)btn * H_;
  for (int t6 = 0; t6 < 6; ++t6) {
    const int ntile = (wave * 6 + t6) * 16;
    v8f acc = {0, 0, 0, 0, 0, 0, 0, 0};
#pragma unroll
    for (int k0 = 0; k0 < L_; k0 += 32) {
      v16bf a = load_a_frag<72>(sAttn, 0, k0, lane);
      v16bf b = load_bt_frag_g(Vt, L_, ntile, k0, lane);
      acc = wmma_bf16(a, b, acc);
    }
    if (lane < 16) {
      const int c = ntile + lane;
      float sum = 0.f;
#pragma unroll
      for (int i = 0; i < 8; ++i) {  // rows 0..7 valid (lanes 0-15)
        mo[(long)i * H_ + c] = acc[i];
        sum += acc[i];
      }
      po[c] = sum * (1.f / E_);
    }
  }
}

// ---------------------------------------------------------------------------
// score[b,t,n] = sigmoid(leaky(wf.Wa_lo + wp.Wa_hi + ba) + dists*wb + bb)
// ---------------------------------------------------------------------------

__global__ void score_kernel(const float* __restrict__ wf, const float* __restrict__ wp,
                             const float* __restrict__ Wa, const float* __restrict__ ba,
                             const float* __restrict__ dists, const float* __restrict__ wb,
                             const float* __restrict__ bb, float* __restrict__ score) {
  const int btn  = blockIdx.x * 8 + (threadIdx.x >> 5);
  const int lane = threadIdx.x & 31;
  const int bt   = btn >> 5;
  float s = 0.f;
  for (int j = lane; j < A_; j += 32) {
    s += wf[bt * A_ + j] * Wa[j];
    s += wp[(long)btn * A_ + j] * Wa[A_ + j];
  }
#pragma unroll
  for (int off = 16; off; off >>= 1) s += __shfl_xor(s, off, 32);
  if (lane == 0) {
    float x = s + ba[0];
    x = (x > 0.f) ? x : 0.01f * x;  // jax leaky_relu slope 0.01
    x += dists[btn] * wb[0] + bb[0];
    score[btn] = 1.f / (1.f + __expf(-x));
  }
}

// delta scatter-add into x (entity spans may overlap -> atomics)
__global__ void scatter_kernel(const float* __restrict__ score, const float* __restrict__ msgs,
                               const int* __restrict__ spans, float* __restrict__ x) {
  const int bte = blockIdx.x;  // 256 = B*T*E
  const int b = bte >> 4, t = (bte >> 3) & 1, e = bte & 7;
  const int bt  = b * T_ + t;
  const int row = b * S_ + spans[bt] + e;
  for (int h = threadIdx.x; h < H_; h += 256) {
    float acc = 0.f;
#pragma unroll 4
    for (int n = 0; n < N_; ++n)
      acc += score[bt * N_ + n] * msgs[((long)(bt * N_ + n) * E_ + e) * H_ + h];
    atomicAdd(&x[(long)row * H_ + h], acc);
  }
}

// ---------------------------------------------------------------------------
// NT GEMM (per batch): C = scale * A[M,K] * B[N,K]^T. Full tiles only.
// Staging via CDNA5 async global->LDS DMA; s_wait_asynccnt 0 before each
// barrier; DMA overlaps the WMMA work of the current k-step.
// ---------------------------------------------------------------------------

__global__ __launch_bounds__(256) void gemm_nt_bf16(
    const bf16* __restrict__ Aall, const bf16* __restrict__ Ball,
    float* __restrict__ Call, int M, int N, int K, float scale) {
  const int bz = blockIdx.z;
  float* C = Call + (long)bz * M * N;
  __shared__ __align__(16) bf16 sA[2][64 * 40];
  __shared__ __align__(16) bf16 sBt[2][64 * 40];
  const int tid = threadIdx.x, lane = tid & 31, wave = tid >> 5;
  const int gm0 = blockIdx.x * 64, gn0 = blockIdx.y * 64;
  const int mt = (wave & 3) * 16, nt = (wave >> 2) * 32;
  const int r = tid >> 2, c = (tid & 3) * 8;
  const bf16* aptr = Aall + (long)bz * M * K + (long)(gm0 + r) * K + c;
  const bf16* bptr = Ball + (long)bz * N * K + (long)(gn0 + r) * K + c;

  auto stage = [&](int buf, int k0) {
    async_g2l_b128(&sA[buf][r * 40 + c],  aptr + k0);
    async_g2l_b128(&sBt[buf][r * 40 + c], bptr + k0);
  };

  v8f acc0 = {0, 0, 0, 0, 0, 0, 0, 0};
  v8f acc1 = {0, 0, 0, 0, 0, 0, 0, 0};
  stage(0, 0);
  wait_async0();
  __syncthreads();
  const int nk = K >> 5;
  for (int s = 0; s < nk; ++s) {
    const int cur = s & 1;
    if (s + 1 < nk) stage(cur ^ 1, (s + 1) << 5);
    v16bf a  = load_a_frag<40>(sA[cur], mt, 0, lane);
    v16bf b0 = load_bt_frag_lds<40>(sBt[cur], nt, 0, lane);
    v16bf b1 = load_bt_frag_lds<40>(sBt[cur], nt + 16, 0, lane);
    acc0 = wmma_bf16(a, b0, acc0);
    acc1 = wmma_bf16(a, b1, acc1);
    wait_async0();
    __syncthreads();
  }
  const int col = lane & 15, rbase = (lane >> 4) * 8;
#pragma unroll
  for (int i = 0; i < 8; ++i) {
    int gr = gm0 + mt + rbase + i;
    C[(long)gr * N + gn0 + nt + col]      = acc0[i] * scale;
    C[(long)gr * N + gn0 + nt + 16 + col] = acc1[i] * scale;
  }
}

// wave-per-row softmax -> bf16
__global__ __launch_bounds__(256) void softmax_rows(const float* __restrict__ logits,
                                                    bf16* __restrict__ attn, int cols) {
  const int row  = blockIdx.x * 8 + (threadIdx.x >> 5);
  const int lane = threadIdx.x & 31;
  const float* p = logits + (long)row * cols;
  float m = -1e30f;
  for (int j = lane; j < cols; j += 32) m = fmaxf(m, p[j]);
#pragma unroll
  for (int off = 16; off; off >>= 1) m = fmaxf(m, __shfl_xor(m, off, 32));
  float s = 0.f;
  for (int j = lane; j < cols; j += 32) s += __expf(p[j] - m);
#pragma unroll
  for (int off = 16; off; off >>= 1) s += __shfl_xor(s, off, 32);
  const float inv = 1.f / s;
  bf16* o = attn + (long)row * cols;
  for (int j = lane; j < cols; j += 32) o[j] = (bf16)(__expf(p[j] - m) * inv);
}

// ---------------------------------------------------------------------------
// NN GEMM (per batch): C_f32 = A_bf16[M,K] * B_bf16[K,N]. Full tiles only.
// A tile staged via async DMA; B tile transposed manually (needs scatter).
// ---------------------------------------------------------------------------

__global__ __launch_bounds__(256) void gemm_nn_bf16(
    const bf16* __restrict__ Aall, const bf16* __restrict__ Ball,
    float* __restrict__ Call, int M, int N, int K) {
  const int bz = blockIdx.z;
  float* C = Call + (long)bz * M * N;
  __shared__ __align__(16) bf16 sA[2][64 * 40];
  __shared__ __align__(16) bf16 sBt[2][64 * 40];
  const int tid = threadIdx.x, lane = tid & 31, wave = tid >> 5;
  const int gm0 = blockIdx.x * 64, gn0 = blockIdx.y * 64;
  const int mt = (wave & 3) * 16, nt = (wave >> 2) * 32;
  const int ar = tid >> 2, ac = (tid & 3) * 8;
  const int bk = tid >> 3, bn = (tid & 7) * 8;
  const bf16* aptr = Aall + (long)bz * M * K + (long)(gm0 + ar) * K + ac;
  const bf16* bptr = Ball + (long)bz * K * N + (long)bk * N + gn0 + bn;

  auto stage = [&](int buf, int k0) {
    async_g2l_b128(&sA[buf][ar * 40 + ac], aptr + k0);
    v8bf v = *(const v8bf*)(bptr + (long)k0 * N);
#pragma unroll
    for (int j = 0; j < 8; ++j) sBt[buf][(bn + j) * 40 + bk] = v[j];
    __builtin_prefetch(bptr + (long)(k0 + 64) * N, 0, 1);
  };

  v8f acc0 = {0, 0, 0, 0, 0, 0, 0, 0};
  v8f acc1 = {0, 0, 0, 0, 0, 0, 0, 0};
  stage(0, 0);
  wait_async0();
  __syncthreads();
  const int nk = K >> 5;
  for (int s = 0; s < nk; ++s) {
    const int cur = s & 1;
    if (s + 1 < nk) stage(cur ^ 1, (s + 1) << 5);
    v16bf a  = load_a_frag<40>(sA[cur], mt, 0, lane);
    v16bf b0 = load_bt_frag_lds<40>(sBt[cur], nt, 0, lane);
    v16bf b1 = load_bt_frag_lds<40>(sBt[cur], nt + 16, 0, lane);
    acc0 = wmma_bf16(a, b0, acc0);
    acc1 = wmma_bf16(a, b1, acc1);
    wait_async0();
    __syncthreads();
  }
  const int col = lane & 15, rbase = (lane >> 4) * 8;
#pragma unroll
  for (int i = 0; i < 8; ++i) {
    int gr = gm0 + mt + rbase + i;
    C[(long)gr * N + gn0 + nt + col]      = acc0[i];
    C[(long)gr * N + gn0 + nt + 16 + col] = acc1[i];
  }
}

// ---------------------------------------------------------------------------
// Host-side orchestration
// ---------------------------------------------------------------------------

static inline char* carve(char** p, size_t bytes) {
  char* r = *p;
  *p += (bytes + 255) & ~(size_t)255;
  return r;
}

extern "C" void kernel_launch(void* const* d_in, const int* in_sizes, int n_in,
                              void* d_out, int out_size, void* d_ws, size_t ws_size,
                              hipStream_t stream) {
  (void)in_sizes; (void)n_in; (void)out_size; (void)ws_size;
  const float* xs    = (const float*)d_in[0];
  const float* neigh = (const float*)d_in[1];
  const float* dists = (const float*)d_in[2];
  const int*   spans = (const int*)d_in[3];
  const float* Wq = (const float*)d_in[4];
  const float* bq = (const float*)d_in[5];
  const float* Wk = (const float*)d_in[6];
  const float* bk = (const float*)d_in[7];
  const float* Wv = (const float*)d_in[8];
  const float* bv = (const float*)d_in[9];
  const float* Ww = (const float*)d_in[10];
  const float* bw = (const float*)d_in[11];
  const float* Wa = (const float*)d_in[12];
  const float* ba = (const float*)d_in[13];
  const float* wb = (const float*)d_in[14];
  const float* bb = (const float*)d_in[15];
  float* out = (float*)d_out;

  char* p = (char*)d_ws;
  bf16*  wq_bf   = (bf16*)carve(&p, (size_t)H_ * D_ * 2);
  bf16*  wk_bf   = (bf16*)carve(&p, (size_t)H_ * D_ * 2);
  bf16*  wv_bf   = (bf16*)carve(&p, (size_t)H_ * H_ * 2);
  bf16*  ww_bf   = (bf16*)carve(&p, (size_t)H_ * A_ * 2);
  bf16*  k_bf    = (bf16*)carve(&p, (size_t)B_ * T_ * N_ * L_ * D_ * 2);
  bf16*  neighT  = (bf16*)carve(&p, (size_t)B_ * T_ * N_ * H_ * L_ * 2);
  float* q_f32   = (float*)carve(&p, (size_t)B_ * T_ * E_ * D_ * 4);
  float* feas    = (float*)carve(&p, (size_t)B_ * T_ * H_ * 4);
  int*   rowidx  = (int*)  carve(&p, (size_t)B_ * T_ * E_ * 4);
  float* msgs    = (float*)carve(&p, (size_t)B_ * T_ * N_ * E_ * H_ * 4);
  float* pooled  = (float*)carve(&p, (size_t)B_ * T_ * N_ * H_ * 4);
  float* wp      = (float*)carve(&p, (size_t)B_ * T_ * N_ * A_ * 4);
  float* wf      = (float*)carve(&p, (size_t)B_ * T_ * A_ * 4);
  float* score   = (float*)carve(&p, (size_t)B_ * T_ * N_ * 4);
  float* x       = (float*)carve(&p, (size_t)B_ * S_ * H_ * 4);
  bf16*  q2_bf   = (bf16*)carve(&p, (size_t)B_ * S_ * D_ * 2);
  bf16*  k2_bf   = (bf16*)carve(&p, (size_t)B_ * S_ * D_ * 2);
  bf16*  v2_bf   = (bf16*)carve(&p, (size_t)B_ * S_ * H_ * 2);
  float* logits2 = (float*)carve(&p, (size_t)B_ * S_ * S_ * 4);
  bf16*  attn_bf = (bf16*)carve(&p, (size_t)B_ * S_ * S_ * 2);

  const float inv_sqrt_d = 0.08838834764831845f;

  // 1. weights -> bf16
  cvt_f32_bf16<<<(H_ * D_ + 255) / 256, 256, 0, stream>>>(Wq, wq_bf, H_ * D_);
  cvt_f32_bf16<<<(H_ * D_ + 255) / 256, 256, 0, stream>>>(Wk, wk_bf, H_ * D_);
  cvt_f32_bf16<<<(H_ * H_ + 255) / 256, 256, 0, stream>>>(Wv, wv_bf, H_ * H_);
  cvt_f32_bf16<<<(H_ * A_ + 255) / 256, 256, 0, stream>>>(Ww, ww_bf, H_ * A_);

  // 2. span helpers
  rowidx_kernel<<<1, 256, 0, stream>>>(spans, rowidx);
  feas_kernel<<<B_ * T_, 256, 0, stream>>>(xs, spans, feas);

  // 3. q = e_fea @ Wq + bq  (gather rows of xs)   M=256
  gemm_a32_wb<<<dim3(4, 2), 256, 0, stream>>>(xs, rowidx, 256, H_, D_, wq_bf, bq,
                                              q_f32, nullptr, nullptr);
  // 4. k = neigh @ Wk + bk; also emits neighT (transposed bf16 neigh)
  gemm_a32_wb<<<dim3(1024, 2), 256, 0, stream>>>(neigh, nullptr, 65536, H_, D_, wk_bf, bk,
                                                 nullptr, k_bf, neighT);

  // 5. per-neighbor SDPA + pooled mean
  neighbor_sdpa<<<B_ * T_ * N_, 256, 0, stream>>>(q_f32, k_bf, neighT, msgs, pooled);

  // 6. wp = pooled @ Ww + bw ; wf = feas @ Ww + bw
  gemm_a32_wb<<<dim3(16, 2), 256, 0, stream>>>(pooled, nullptr, 1024, H_, A_, ww_bf, bw,
                                               wp, nullptr, nullptr);
  gemm_a32_wb<<<dim3(1, 2), 256, 0, stream>>>(feas, nullptr, 32, H_, A_, ww_bf, bw,
                                              wf, nullptr, nullptr);

  // 7. scores + scatter-add into x
  score_kernel<<<(B_ * T_ * N_) / 8, 256, 0, stream>>>(wf, wp, Wa, ba, dists, wb, bb, score);
  copy_f32<<<(B_ * S_ * H_ + 255) / 256, 256, 0, stream>>>(xs, x, (long)B_ * S_ * H_);
  scatter_kernel<<<B_ * T_ * E_, 256, 0, stream>>>(score, msgs, spans, x);

  // 8. final attention projections (M = B*S = 8192)
  gemm_a32_wb<<<dim3(128, 2), 256, 0, stream>>>(x, nullptr, 8192, H_, D_, wq_bf, bq,
                                                nullptr, q2_bf, nullptr);
  gemm_a32_wb<<<dim3(128, 2), 256, 0, stream>>>(x, nullptr, 8192, H_, D_, wk_bf, bk,
                                                nullptr, k2_bf, nullptr);
  gemm_a32_wb<<<dim3(128, 12), 256, 0, stream>>>(x, nullptr, 8192, H_, H_, wv_bf, bv,
                                                 nullptr, v2_bf, nullptr);

  // 9. s2 = q2 k2^T / sqrt(D); softmax; out = attn @ v2
  gemm_nt_bf16<<<dim3(8, 8, B_), 256, 0, stream>>>(q2_bf, k2_bf, logits2, S_, S_, D_, inv_sqrt_d);
  softmax_rows<<<(B_ * S_) / 8, 256, 0, stream>>>(logits2, attn_bf, S_);
  gemm_nn_bf16<<<dim3(8, 12, B_), 256, 0, stream>>>(attn_bf, v2_bf, out, S_, H_, S_);
}
